// Longformer_13743895347225
// MI455X (gfx1250) — compile-verified, matrix operand order checked by reading, not run on hardware
//
#include <hip/hip_runtime.h>
#include <cstdint>
#include <cstddef>

// ---------------- types ----------------
typedef __bf16 bf16_t;
typedef __attribute__((ext_vector_type(16))) __bf16 v16bf;
typedef __attribute__((ext_vector_type(8)))  __bf16 v8bf;
typedef __attribute__((ext_vector_type(8)))  float  v8f;

// ---------------- model constants ----------------
#define S_LEN   4096
#define DMODEL  768
#define NHEAD   12
#define DHEAD   64
#define NLAYER  12
#define FFDIM   3072
#define WWIN    128
#define NBLK    32          // S_LEN / WWIN
#define KB3     384         // 3 * WWIN
#define SCOLS   392         // padded row stride (385 -> 392, keeps 16B alignment for bf16 rows)
#define VTST    392         // V-transpose row stride

static __device__ __forceinline__ float bf2f(bf16_t x) { return (float)x; }
static __device__ __forceinline__ bf16_t f2bf(float x) { return (bf16_t)x; }

// Load a 16-element bf16 fragment per the CDNA5 16-bit A/B layout:
// elems 0..7 = contiguous K at p, elems 8..15 = contiguous K at p+16.
static __device__ __forceinline__ v16bf load_frag(const bf16_t* p) {
    v8bf lo = *(const v8bf*)p;
    v8bf hi = *(const v8bf*)(p + 16);
    v16bf r;
#pragma unroll
    for (int i = 0; i < 8; i++) { r[i] = lo[i]; r[i + 8] = hi[i]; }
    return r;
}

static __device__ __forceinline__ v8f wmma_bf16(v16bf a, v16bf b, v8f c) {
    return __builtin_amdgcn_wmma_f32_16x16x32_bf16(false, a, false, b, (short)0, c, false, false);
}

// ---------------- fp32 -> bf16 transpose (weights) ----------------
// in:  W [K][N] fp32 row-major.  out: WT [N][K] bf16 row-major.
__global__ void convT_kernel(const float* __restrict__ W, bf16_t* __restrict__ WT,
                             int K, int N) {
    size_t t = (size_t)blockIdx.x * blockDim.x + threadIdx.x;
    size_t total = (size_t)K * N;
    if (t >= total) return;
    int k = (int)(t / N);
    int n = (int)(t % N);
    WT[(size_t)n * K + k] = f2bf(W[t]);
}

// ---------------- embedding gather ----------------
__global__ void embed_kernel(const int* __restrict__ ids,
                             const float* __restrict__ wemb,
                             const float* __restrict__ pemb,
                             float* __restrict__ hpre,
                             float* __restrict__ maskf) {
    int s = blockIdx.x;
    int id = ids[s];
    if (threadIdx.x == 0) maskf[s] = (id != 1) ? 1.0f : 0.0f;  // PAD == 1
    const float* wr = wemb + (size_t)id * DMODEL;
    const float* pr = pemb + (size_t)s * DMODEL;
    float* o = hpre + (size_t)s * DMODEL;
    for (int d = threadIdx.x; d < DMODEL; d += blockDim.x) o[d] = wr[d] + pr[d];
}

// ---------------- residual add + LayerNorm (writes fp32 + bf16) ----------------
__global__ __launch_bounds__(256)
void add_ln_kernel(const float* __restrict__ a, const float* __restrict__ b,
                   const float* __restrict__ gamma, const float* __restrict__ beta,
                   float* __restrict__ outF, bf16_t* __restrict__ outB) {
    __shared__ float red[256];
    __shared__ float red2[256];
    __shared__ float mv[2];
    int s = blockIdx.x;
    const float* ar = a + (size_t)s * DMODEL;
    const float* br = b ? (b + (size_t)s * DMODEL) : nullptr;
    float x[3];
    float sum = 0.f, sq = 0.f;
#pragma unroll
    for (int i = 0; i < 3; i++) {
        int d = threadIdx.x + i * 256;
        float v = ar[d] + (br ? br[d] : 0.0f);
        x[i] = v; sum += v; sq += v * v;
    }
    red[threadIdx.x] = sum; red2[threadIdx.x] = sq;
    __syncthreads();
    for (int off = 128; off > 0; off >>= 1) {
        if (threadIdx.x < off) {
            red[threadIdx.x]  += red[threadIdx.x + off];
            red2[threadIdx.x] += red2[threadIdx.x + off];
        }
        __syncthreads();
    }
    if (threadIdx.x == 0) {
        float m = red[0] / (float)DMODEL;
        float var = red2[0] / (float)DMODEL - m * m;
        mv[0] = m; mv[1] = rsqrtf(var + 1e-5f);
    }
    __syncthreads();
    float m = mv[0], rs = mv[1];
#pragma unroll
    for (int i = 0; i < 3; i++) {
        int d = threadIdx.x + i * 256;
        float v = (x[i] - m) * rs * gamma[d] + beta[d];
        size_t idx = (size_t)s * DMODEL + d;
        outF[idx] = v;
        outB[idx] = f2bf(v);
    }
}

// ---------------- generic bf16 WMMA GEMM ----------------
// C[M][N] = act( (A[M][K] * BT[N][K]^T + bias) * scale ), 4 waves/block,
// each wave computes 32x64.  grid = (M/128, N/64).
__global__ __launch_bounds__(128)
void gemm_bf16_kernel(const bf16_t* __restrict__ A, const bf16_t* __restrict__ BT,
                      const float* __restrict__ bias,
                      float* __restrict__ outF, bf16_t* __restrict__ outB,
                      int M, int N, int K, float scale, int gelu_flag) {
    const int lane = threadIdx.x & 31;
    const int wave = threadIdx.x >> 5;
    const int l16  = lane & 15;
    const int koff = (lane >> 4) * 8;
    const int rowHi = (lane >> 4) * 8;
    const int mBase = blockIdx.x * 128 + wave * 32;
    const int nBase = blockIdx.y * 64;
    (void)M;

    v8f acc[2][4] = {};

    for (int kk = 0; kk < K; kk += 32) {
        v16bf a0 = load_frag(A + (size_t)(mBase + l16) * K + kk + koff);
        v16bf a1 = load_frag(A + (size_t)(mBase + 16 + l16) * K + kk + koff);
#pragma unroll
        for (int nt = 0; nt < 4; nt++) {
            v16bf b = load_frag(BT + (size_t)(nBase + nt * 16 + l16) * K + kk + koff);
            acc[0][nt] = wmma_bf16(a0, b, acc[0][nt]);
            acc[1][nt] = wmma_bf16(a1, b, acc[1][nt]);
        }
    }

#pragma unroll
    for (int ms = 0; ms < 2; ms++) {
#pragma unroll
        for (int nt = 0; nt < 4; nt++) {
            int col = nBase + nt * 16 + l16;
            float bc = bias ? bias[col] : 0.0f;
#pragma unroll
            for (int r = 0; r < 8; r++) {
                int row = mBase + ms * 16 + r + rowHi;
                float v = (acc[ms][nt][r] + bc) * scale;
                if (gelu_flag) {
                    float t = tanhf(0.7978845608028654f * (v + 0.044715f * v * v * v));
                    v = 0.5f * v * (1.0f + t);
                }
                size_t idx = (size_t)row * N + col;
                if (outF) outF[idx] = v;
                if (outB) outB[idx] = f2bf(v);
            }
        }
    }
}

// ---------------- banded (sliding-window) attention ----------------
// grid = NHEAD*NBLK blocks of 256 threads (8 waves); wave w handles 16 queries.
// LDS: [0, 128*392*4)            fp32 scores (later reused as V^T bf16)
//      [SCF_B, SCF_B + 128*392*2) bf16 probs
//      [+..., +256)                v0 (global value row) fp32
#define SCF_BYTES  (128 * SCOLS * 4)
#define PRB_BYTES  (128 * SCOLS * 2)
#define BANDED_LDS (SCF_BYTES + PRB_BYTES + DHEAD * 4)

__global__ __launch_bounds__(256)
void banded_attn_kernel(const bf16_t* __restrict__ Q, const bf16_t* __restrict__ Kb,
                        const bf16_t* __restrict__ Vb, const float* __restrict__ maskf,
                        bf16_t* __restrict__ Ob) {
    extern __shared__ char smem[];
    float*  scF = (float*)smem;
    bf16_t* prB = (bf16_t*)(smem + SCF_BYTES);
    float*  v0f = (float*)(smem + SCF_BYTES + PRB_BYTES);
    bf16_t* VT  = (bf16_t*)smem;   // reuses score region after softmax

    const int h    = blockIdx.x / NBLK;
    const int nb   = blockIdx.x % NBLK;
    const int lane = threadIdx.x & 31;
    const int wave = threadIdx.x >> 5;
    const int l16  = lane & 15;
    const int koff = (lane >> 4) * 8;
    const int rowHi = (lane >> 4) * 8;
    const size_t hoff = (size_t)h * DHEAD;

    // ---- phase 1: banded scores via WMMA (Q block 16xDH  x  K^T DHx16-keys) ----
    {
        const int qrow = nb * WWIN + wave * 16 + l16;
        const bf16_t* qp = Q + (size_t)qrow * DMODEL + hoff;
        v16bf a0 = load_frag(qp + 0  + koff);
        v16bf a1 = load_frag(qp + 32 + koff);
        for (int nt = 0; nt < 24; nt++) {
            int keyIdx = nt * 16 + l16;               // this lane's key column
            int keyAbs = nb * WWIN - WWIN + keyIdx;
            int keyClamped = min(max(keyAbs, 0), S_LEN - 1);
            const bf16_t* kp = Kb + (size_t)keyClamped * DMODEL + hoff;
            v16bf b0 = load_frag(kp + 0  + koff);
            v16bf b1 = load_frag(kp + 32 + koff);
            v8f acc = {};
            acc = wmma_bf16(a0, b0, acc);
            acc = wmma_bf16(a1, b1, acc);
            bool kvalid = (keyAbs >= 1) && (keyAbs < S_LEN) && (maskf[keyClamped] > 0.5f);
#pragma unroll
            for (int r = 0; r < 8; r++) {
                int q16  = r + rowHi;
                int qAbs = nb * WWIN + wave * 16 + q16;
                int rel  = keyAbs - qAbs;
                bool ok  = kvalid && (rel >= -WWIN) && (rel <= WWIN);
                scF[(wave * 16 + q16) * SCOLS + keyIdx] = ok ? acc[r] : -1e9f;
            }
        }
    }
    // global-key column (index 384) + stage v0
    if (threadIdx.x < 128) {
        int qAbs = nb * WWIN + threadIdx.x;
        const bf16_t* qp = Q + (size_t)qAbs * DMODEL + hoff;
        const bf16_t* kp = Kb + hoff;           // key row 0
        float sg = 0.f;
#pragma unroll 8
        for (int d = 0; d < DHEAD; d++) sg += bf2f(qp[d]) * bf2f(kp[d]);
        scF[threadIdx.x * SCOLS + 384] = sg;
    }
    if (threadIdx.x < DHEAD) v0f[threadIdx.x] = bf2f(Vb[hoff + threadIdx.x]);
    __syncthreads();

    // ---- phase 2: softmax over 385 values per query row -> bf16 probs ----
    if (threadIdx.x < 128) {
        float* row = scF + threadIdx.x * SCOLS;
        float mx = -3.4e38f;
        for (int j = 0; j < 385; j++) mx = fmaxf(mx, row[j]);
        float sum = 0.f;
        for (int j = 0; j < 385; j++) { float e = __expf(row[j] - mx); row[j] = e; sum += e; }
        float inv = 1.0f / sum;
        bf16_t* prow = prB + threadIdx.x * SCOLS;
        for (int j = 0; j < 385; j++) prow[j] = f2bf(row[j] * inv);
        // zero the pad columns so WMMA reads are clean
        for (int j = 385; j < SCOLS; j++) prow[j] = f2bf(0.0f);
    }
    __syncthreads();

    // ---- stage V^T into (dead) score region: VT[dh][key], bf16 ----
    for (int u = threadIdx.x; u < KB3 * 8; u += 256) {
        int key = u >> 3;
        int dc  = (u & 7) * 8;
        int keyAbs = nb * WWIN - WWIN + key;
        v8bf val;
        if (keyAbs >= 0 && keyAbs < S_LEN) {
            val = *(const v8bf*)(Vb + (size_t)keyAbs * DMODEL + hoff + dc);
        } else {
#pragma unroll
            for (int i = 0; i < 8; i++) val[i] = f2bf(0.0f);
        }
#pragma unroll
        for (int i = 0; i < 8; i++) VT[(size_t)(dc + i) * VTST + key] = val[i];
    }
    __syncthreads();

    // ---- phase 3: out = P[16x384] x V[384x64] + p_g * v0, via WMMA from LDS ----
    {
        v8f acc[4] = {};
        const bf16_t* prow = prB + (size_t)(wave * 16 + l16) * SCOLS;
        for (int kk = 0; kk < KB3; kk += 32) {
            v16bf a = load_frag(prow + kk + koff);
#pragma unroll
            for (int nt = 0; nt < 4; nt++) {
                v16bf b = load_frag(VT + (size_t)(nt * 16 + l16) * VTST + kk + koff);
                acc[nt] = wmma_bf16(a, b, acc[nt]);
            }
        }
#pragma unroll
        for (int nt = 0; nt < 4; nt++) {
            int dh = nt * 16 + l16;
            float v0v = v0f[dh];
#pragma unroll
            for (int r = 0; r < 8; r++) {
                int q16 = r + rowHi;
                int qLocal = wave * 16 + q16;
                float pg = bf2f(prB[qLocal * SCOLS + 384]);
                int qAbs = nb * WWIN + qLocal;
                Ob[(size_t)qAbs * DMODEL + hoff + dh] = f2bf(acc[nt][r] + pg * v0v);
            }
        }
    }
}

// ---------------- global query (token 0 attends to all non-pad keys) ----------------
__global__ __launch_bounds__(256)
void global_q_kernel(const bf16_t* __restrict__ Q, const bf16_t* __restrict__ Kb,
                     const bf16_t* __restrict__ Vb, const float* __restrict__ maskf,
                     bf16_t* __restrict__ Ob, float* __restrict__ scratch) {
    __shared__ float q0[DHEAD];
    __shared__ float red[256];
    __shared__ float ogs[DHEAD];
    int h = blockIdx.x;
    size_t hoff = (size_t)h * DHEAD;
    if (threadIdx.x < DHEAD) { q0[threadIdx.x] = bf2f(Q[hoff + threadIdx.x]); ogs[threadIdx.x] = 0.f; }
    __syncthreads();
    float* sc = scratch + (size_t)h * S_LEN;
    float mx = -3.4e38f;
    for (int s = threadIdx.x; s < S_LEN; s += 256) {
        const bf16_t* kp = Kb + (size_t)s * DMODEL + hoff;
        float d = 0.f;
#pragma unroll 8
        for (int j = 0; j < DHEAD; j++) d += q0[j] * bf2f(kp[j]);
        if (maskf[s] < 0.5f) d = -1e9f;
        sc[s] = d;
        mx = fmaxf(mx, d);
    }
    red[threadIdx.x] = mx; __syncthreads();
    for (int off = 128; off > 0; off >>= 1) {
        if (threadIdx.x < off) red[threadIdx.x] = fmaxf(red[threadIdx.x], red[threadIdx.x + off]);
        __syncthreads();
    }
    float gmx = red[0];
    __syncthreads();
    float lsum = 0.f;
    float og[DHEAD];
#pragma unroll
    for (int j = 0; j < DHEAD; j++) og[j] = 0.f;
    for (int s = threadIdx.x; s < S_LEN; s += 256) {
        float e = __expf(sc[s] - gmx);
        lsum += e;
        const bf16_t* vp = Vb + (size_t)s * DMODEL + hoff;
#pragma unroll 8
        for (int j = 0; j < DHEAD; j++) og[j] += e * bf2f(vp[j]);
    }
    red[threadIdx.x] = lsum; __syncthreads();
    for (int off = 128; off > 0; off >>= 1) {
        if (threadIdx.x < off) red[threadIdx.x] += red[threadIdx.x + off];
        __syncthreads();
    }
    float tot = red[0];
#pragma unroll
    for (int j = 0; j < DHEAD; j++) atomicAdd(&ogs[j], og[j]);
    __syncthreads();
    if (threadIdx.x < DHEAD) Ob[hoff + threadIdx.x] = f2bf(ogs[threadIdx.x] / tot);
}

// ---------------- final copy ----------------
__global__ void copy_kernel(const float* __restrict__ in, float* __restrict__ out, size_t n) {
    size_t t = (size_t)blockIdx.x * blockDim.x + threadIdx.x;
    if (t < n) out[t] = in[t];
}

// ================= host-side launch =================
extern "C" void kernel_launch(void* const* d_in, const int* in_sizes, int n_in,
                              void* d_out, int out_size, void* d_ws, size_t ws_size,
                              hipStream_t stream) {
    (void)in_sizes; (void)n_in; (void)out_size; (void)ws_size;

    const int*   input_ids = (const int*)d_in[0];
    const float* word_emb  = (const float*)d_in[1];
    const float* pos_emb   = (const float*)d_in[2];
    const float* emb_ln_s  = (const float*)d_in[3];
    const float* emb_ln_b  = (const float*)d_in[4];
    const float* Wq = (const float*)d_in[5];
    const float* bq = (const float*)d_in[6];
    const float* Wk = (const float*)d_in[7];
    const float* bk = (const float*)d_in[8];
    const float* Wv = (const float*)d_in[9];
    const float* bv = (const float*)d_in[10];
    const float* Wo = (const float*)d_in[11];
    const float* bo = (const float*)d_in[12];
    const float* ln1_s = (const float*)d_in[13];
    const float* ln1_b = (const float*)d_in[14];
    const float* W1 = (const float*)d_in[15];
    const float* b1 = (const float*)d_in[16];
    const float* W2 = (const float*)d_in[17];
    const float* b2 = (const float*)d_in[18];
    const float* ln2_s = (const float*)d_in[19];
    const float* ln2_b = (const float*)d_in[20];

    const size_t SD = (size_t)S_LEN * DMODEL;
    const size_t SF = (size_t)S_LEN * FFDIM;
    const size_t DD = (size_t)DMODEL * DMODEL;
    const size_t DF = (size_t)DMODEL * FFDIM;

    char* wsb = (char*)d_ws;
    size_t off = 0;
    auto take = [&](size_t bytes) -> char* {
        char* p = wsb + off;
        off = (off + bytes + 255) & ~(size_t)255;
        return p;
    };
    float*  hA    = (float*)take(SD * 4);       // hidden (residual A)
    float*  hB    = (float*)take(SD * 4);       // hidden (residual B)
    float*  projF = (float*)take(SD * 4);       // fp32 GEMM out / embed pre-LN
    bf16_t* hb    = (bf16_t*)take(SD * 2);      // current hidden bf16 (GEMM A)
    bf16_t* ob    = (bf16_t*)take(SD * 2);      // attention output bf16
    bf16_t* qb    = (bf16_t*)take(SD * 2);
    bf16_t* kb    = (bf16_t*)take(SD * 2);
    bf16_t* vb    = (bf16_t*)take(SD * 2);
    bf16_t* f1b   = (bf16_t*)take(SF * 2);      // FFN intermediate bf16
    bf16_t* wqT   = (bf16_t*)take(DD * 2);
    bf16_t* wkT   = (bf16_t*)take(DD * 2);
    bf16_t* wvT   = (bf16_t*)take(DD * 2);
    bf16_t* woT   = (bf16_t*)take(DD * 2);
    bf16_t* w1T   = (bf16_t*)take(DF * 2);
    bf16_t* w2T   = (bf16_t*)take(DF * 2);
    float*  maskf = (float*)take((size_t)S_LEN * 4);
    float*  gqScr = (float*)take((size_t)NHEAD * S_LEN * 4);

    // allow the large dynamic-LDS attention kernel (CDNA5: 320 KB per WGP)
    (void)hipFuncSetAttribute((const void*)banded_attn_kernel,
                              hipFuncAttributeMaxDynamicSharedMemorySize, BANDED_LDS);

    auto g1 = [](size_t n) { return dim3((unsigned)((n + 255) / 256)); };

    // embedding + embed LayerNorm
    embed_kernel<<<S_LEN, 256, 0, stream>>>(input_ids, word_emb, pos_emb, projF, maskf);
    add_ln_kernel<<<S_LEN, 256, 0, stream>>>(projF, nullptr, emb_ln_s, emb_ln_b, hA, hb);

    const dim3 gridD(32, DMODEL / 64);   // M/128 x N/64 for N=768
    const dim3 gridF(32, FFDIM / 64);    // N=3072

    for (int l = 0; l < NLAYER; l++) {
        // convert + transpose this layer's weights to bf16
        convT_kernel<<<g1(DD), 256, 0, stream>>>(Wq + (size_t)l * DD, wqT, DMODEL, DMODEL);
        convT_kernel<<<g1(DD), 256, 0, stream>>>(Wk + (size_t)l * DD, wkT, DMODEL, DMODEL);
        convT_kernel<<<g1(DD), 256, 0, stream>>>(Wv + (size_t)l * DD, wvT, DMODEL, DMODEL);
        convT_kernel<<<g1(DD), 256, 0, stream>>>(Wo + (size_t)l * DD, woT, DMODEL, DMODEL);
        convT_kernel<<<g1(DF), 256, 0, stream>>>(W1 + (size_t)l * DF, w1T, DMODEL, FFDIM);
        convT_kernel<<<g1(DF), 256, 0, stream>>>(W2 + (size_t)l * DF, w2T, FFDIM, DMODEL);

        // Q (scaled by DH^-0.5), K, V projections -> bf16
        gemm_bf16_kernel<<<gridD, 128, 0, stream>>>(hb, wqT, bq + (size_t)l * DMODEL,
                                                    nullptr, qb, S_LEN, DMODEL, DMODEL, 0.125f, 0);
        gemm_bf16_kernel<<<gridD, 128, 0, stream>>>(hb, wkT, bk + (size_t)l * DMODEL,
                                                    nullptr, kb, S_LEN, DMODEL, DMODEL, 1.0f, 0);
        gemm_bf16_kernel<<<gridD, 128, 0, stream>>>(hb, wvT, bv + (size_t)l * DMODEL,
                                                    nullptr, vb, S_LEN, DMODEL, DMODEL, 1.0f, 0);

        // attention
        banded_attn_kernel<<<NHEAD * NBLK, 256, BANDED_LDS, stream>>>(qb, kb, vb, maskf, ob);
        global_q_kernel<<<NHEAD, 256, 0, stream>>>(qb, kb, vb, maskf, ob, gqScr);

        // output projection + residual + LN1
        gemm_bf16_kernel<<<gridD, 128, 0, stream>>>(ob, woT, bo + (size_t)l * DMODEL,
                                                    projF, nullptr, S_LEN, DMODEL, DMODEL, 1.0f, 0);
        add_ln_kernel<<<S_LEN, 256, 0, stream>>>(hA, projF, ln1_s + (size_t)l * DMODEL,
                                                 ln1_b + (size_t)l * DMODEL, hB, hb);

        // FFN: gelu(h2 @ W1 + b1) @ W2 + b2, residual + LN2
        gemm_bf16_kernel<<<gridF, 128, 0, stream>>>(hb, w1T, b1 + (size_t)l * FFDIM,
                                                    nullptr, f1b, S_LEN, FFDIM, DMODEL, 1.0f, 1);
        gemm_bf16_kernel<<<gridD, 128, 0, stream>>>(f1b, w2T, b2 + (size_t)l * DMODEL,
                                                    projF, nullptr, S_LEN, DMODEL, FFDIM, 1.0f, 0);
        add_ln_kernel<<<S_LEN, 256, 0, stream>>>(hB, projF, ln2_s + (size_t)l * DMODEL,
                                                 ln2_b + (size_t)l * DMODEL, hA, hb);
    }

    copy_kernel<<<g1(SD), 256, 0, stream>>>(hA, (float*)d_out, SD);
}
